// attn_layer_33397665693756
// MI455X (gfx1250) — matrix operand order repaired and checked by reading, not compiled, not run on hardware
//
#include <hip/hip_runtime.h>

// ---------------------------------------------------------------------------
// CDNA5 (gfx1250) bf16-WMMA flash-attention, TDM-fed double-buffered LDS:
//   Qx = x@Q^T [b,n,64]; Kx = x@K^T [b,n,64]; Vx = x@V^T [b,n,256]
//   S[j,i] = Qx[j].Kx[i]; P = softmax_i(S); out[j] = sum_i P[j,i]*Vx[i]
// b=4, n=4096, d=256, a=64. One-pass online softmax, f32 accumulation.
// ---------------------------------------------------------------------------

typedef __bf16 bf16;
typedef __attribute__((ext_vector_type(16))) __bf16 v16bf;
typedef __attribute__((ext_vector_type(8)))  __bf16 v8bf;
typedef __attribute__((ext_vector_type(8)))  float  v8f;
typedef __attribute__((ext_vector_type(4)))  unsigned int u32x4;
typedef __attribute__((ext_vector_type(8)))  int          i32x8;
typedef __attribute__((ext_vector_type(4)))  int          i32x4;

#define B_SZ   4
#define NTOK   4096
#define DDIM   256
#define ADIM   64

static __device__ inline v8f wmma_bf16(v16bf a, v16bf b, v8f c) {
  // D = A(16x32 bf16) * B(32x16 bf16) + C(16x16 f32)
  return __builtin_amdgcn_wmma_f32_16x16x32_bf16(
      /*neg_a=*/false, a, /*neg_b=*/false, b,
      /*c_mod=*/(short)0, c, /*reuse_a=*/false, /*reuse_b=*/false);
}

// A-operand (16x32, MxK) from row-major bf16 [16 rows x ld]:
// lane L: row = L&15, holds K = kb..kb+15, kb = 16*(L>>4)
static __device__ inline v16bf frag_a_bf16(const bf16* base, int ld, int lane) {
  const int row = lane & 15;
  const int kb  = (lane >> 4) << 4;
  const bf16* p = base + row * ld + kb;
  v8bf lo = *(const v8bf*)(p);
  v8bf hi = *(const v8bf*)(p + 8);
  return __builtin_shufflevector(lo, hi, 0,1,2,3,4,5,6,7,8,9,10,11,12,13,14,15);
}

// B-operand (32x16, KxN) from B^T stored row-major (N rows x ld over K):
// lane L: col = L&15, holds K = kb..kb+15, kb = 16*(L>>4)
static __device__ inline v16bf frag_b_bf16(const bf16* base, int ld, int lane) {
  const int col = lane & 15;
  const int kb  = (lane >> 4) << 4;
  const bf16* p = base + col * ld + kb;
  v8bf lo = *(const v8bf*)(p);
  v8bf hi = *(const v8bf*)(p + 8);
  return __builtin_shufflevector(lo, hi, 0,1,2,3,4,5,6,7,8,9,10,11,12,13,14,15);
}

// Same layouts, converting from f32 source on the fly (stage-1 only).
static __device__ inline v16bf frag_a_f32(const float* base, int ld, int lane) {
  const int row = lane & 15;
  const int kb  = (lane >> 4) << 4;
  const float* p = base + row * ld + kb;
  v16bf r;
#pragma unroll
  for (int i = 0; i < 8; ++i) { r[i] = (bf16)p[i]; r[i + 8] = (bf16)p[i + 8]; }
  return r;
}
static __device__ inline v16bf frag_b_f32(const float* base, int ld, int lane) {
  const int col = lane & 15;
  const int kb  = (lane >> 4) << 4;
  const float* p = base + col * ld + kb;
  v16bf r;
#pragma unroll
  for (int i = 0; i < 16; ++i) r[i] = (bf16)p[i];
  return r;
}

// ---------------------------------------------------------------------------
// Tensor Data Mover: 2D tile load Global -> LDS (bf16 elements).
// D# layout per CDNA5 ISA ch.8:
//   group0: [1:0]=count=1, [63:32]=lds_addr, [120:64]=global_addr, [127:126]=2
//   group1: [17:16]=data_size(1=2B), [79:48]=tensor_dim0, [111:80]=tensor_dim1,
//           [127:112]=tile_dim0, [143:128]=tile_dim1, [207:160]=dim0_stride
//   groups 2/3 = 0 (<=2D tensor). LDS result is fully packed row-major
//   [tile_y][tile_x] starting at lds_addr.
// ---------------------------------------------------------------------------
static __device__ inline unsigned lds_offset(const void* p) {
  return (unsigned)(uintptr_t)p;  // LDS aperture: addr[31:0] = LDS byte offset
}

static __device__ inline void tdm_load_2d_bf16(unsigned lds_off, const void* gptr,
                                               unsigned tile_x, unsigned tile_y,
                                               unsigned tensor_x, unsigned tensor_y,
                                               unsigned row_stride_elems) {
  const unsigned long long ga = (unsigned long long)(uintptr_t)gptr;
  u32x4 g0;
  g0[0] = 1u;                                    // count=1, user descriptor
  g0[1] = lds_off;                               // lds_addr (bytes)
  g0[2] = (unsigned)ga;                          // global_addr[31:0]
  g0[3] = (unsigned)((ga >> 32) & 0x01ffffffu)   // global_addr[56:32]
        | (2u << 30);                            // type=2 ("image")
  i32x8 g1;
  g1[0] = (int)(1u << 16);                       // data_size = 2 bytes
  g1[1] = (int)((tensor_x & 0xffffu) << 16);                                 // dim0[15:0]
  g1[2] = (int)(((tensor_x >> 16) & 0xffffu) | ((tensor_y & 0xffffu) << 16));// dim0[31:16]|dim1[15:0]
  g1[3] = (int)(((tensor_y >> 16) & 0xffffu) | ((tile_x  & 0xffffu) << 16)); // dim1[31:16]|tile0
  g1[4] = (int)(tile_y & 0xffffu);               // tile_dim1 (tile_dim2 = 0)
  g1[5] = (int)row_stride_elems;                 // tensor_dim0_stride[31:0]
  g1[6] = 0;
  g1[7] = 0;
  const i32x4 z4 = {0, 0, 0, 0};
#if defined(__clang_major__) && (__clang_major__ >= 23)
  const i32x8 z8 = {0, 0, 0, 0, 0, 0, 0, 0};
  __builtin_amdgcn_tensor_load_to_lds(g0, g1, z4, z4, z8, 0);
#else
  __builtin_amdgcn_tensor_load_to_lds(g0, g1, z4, z4, 0);
#endif
}

// ---------------------------------------------------------------------------
// Stage 1: projection GEMM  out = bf16( X[Mx256] @ W[Nx256]^T )
//   transposed==0: out row-major [M][N]
//   transposed==1: out as [batch][N][4096] (per-batch transpose, for V^T)
// One wave per 16x16 tile.
// ---------------------------------------------------------------------------
__global__ __launch_bounds__(128)
void proj_bf16_wmma(const float* __restrict__ X, const float* __restrict__ W,
                    bf16* __restrict__ out, int M, int N, int transposed) {
  const int Kd   = DDIM;
  const int lane = threadIdx.x & 31;
  const int wid  = blockIdx.x * 4 + (threadIdx.x >> 5);
  const int tiles_n = N >> 4;
  const int tm = wid / tiles_n;           // wave-uniform -> EXEC stays full
  const int tn = wid - tm * tiles_n;
  if (tm >= (M >> 4)) return;

  const float* xb = X + (size_t)(tm * 16) * Kd;
  const float* wb = W + (size_t)(tn * 16) * Kd;

  v8f acc = {0.f, 0.f, 0.f, 0.f, 0.f, 0.f, 0.f, 0.f};
#pragma unroll
  for (int k = 0; k < Kd; k += 32) {
    v16bf a = frag_a_f32(xb + k, Kd, lane);
    v16bf b = frag_b_f32(wb + k, Kd, lane);
    acc = wmma_bf16(a, b, acc);
  }

  const int mloc = (lane >> 4) << 3;       // C/D layout: M = r + 8*(lane>>4)
  const int n_g  = tn * 16 + (lane & 15);  //             N = lane&15
  if (!transposed) {
#pragma unroll
    for (int r = 0; r < 8; ++r) {
      const int m_g = tm * 16 + mloc + r;
      out[(size_t)m_g * N + n_g] = (bf16)acc[r];
    }
  } else {
#pragma unroll
    for (int r = 0; r < 8; ++r) {
      const int m_g  = tm * 16 + mloc + r;
      const int bidx = m_g >> 12;          // / 4096
      const int tok  = m_g & (NTOK - 1);
      out[((size_t)(bidx * N + n_g)) * NTOK + tok] = (bf16)acc[r];
    }
  }
}

// ---------------------------------------------------------------------------
// Stage 2: TDM-fed flash attention. Block = 4 waves x 16 queries; grid =
// (4096/64, 4). Per 32-key step: wave0 DMAs next K tile (32x64) and V tile
// (256x32, from V^T so LDS rows are B-fragments) into the spare LDS buffer
// via tensor_load_to_lds while all waves run 20 WMMAs on the current buffer.
// Online softmax via 16-lane xor-shuffle reductions; P transposed to the
// WMMA A-layout through a wave-private LDS slab (LDS ops in-order per wave).
// ---------------------------------------------------------------------------
__global__ __launch_bounds__(128)
void flash_attn_wmma(const bf16* __restrict__ Qx,  // [B][NTOK][64]
                     const bf16* __restrict__ Kx,  // [B][NTOK][64]
                     const bf16* __restrict__ Vt,  // [B][256][NTOK]
                     float* __restrict__ out) {    // [B][NTOK][256]
  __shared__ __align__(16) bf16 lds_k[2][32 * ADIM];  // 2 x 4 KB  [key][a]
  __shared__ __align__(16) bf16 lds_v[2][DDIM * 32];  // 2 x 16 KB [e][key]
  __shared__ __align__(16) bf16 lds_p[4][16 * 32];    // per-wave P tile

  const int lane = threadIdx.x & 31;
  const int wv   = threadIdx.x >> 5;
  const int b    = blockIdx.y;
  const int q0   = (blockIdx.x * 4 + wv) * 16;
  const bool issuer = (wv == 0);           // wave-uniform branch

  const bf16* qbase = Qx + ((size_t)b * NTOK + q0) * ADIM;
  const bf16* kbase = Kx + (size_t)b * NTOK * ADIM;
  const bf16* vbase = Vt + (size_t)b * DDIM * NTOK;

  // Prologue: DMA tile 0 into buffer 0.
  if (issuer) {
    tdm_load_2d_bf16(lds_offset(&lds_k[0][0]), kbase,
                     /*tile*/ ADIM, 32, /*tensor*/ ADIM, NTOK, /*stride*/ ADIM);
    tdm_load_2d_bf16(lds_offset(&lds_v[0][0]), vbase,
                     /*tile*/ 32, DDIM, /*tensor*/ NTOK, DDIM, /*stride*/ NTOK);
  }

  // Query A-fragments (a = 0..31 and 32..63), reused for all key tiles.
  const v16bf aq0 = frag_a_bf16(qbase,      ADIM, lane);
  const v16bf aq1 = frag_a_bf16(qbase + 32, ADIM, lane);

  v8f o[16];
#pragma unroll
  for (int t = 0; t < 16; ++t)
#pragma unroll
    for (int r = 0; r < 8; ++r) o[t][r] = 0.f;

  float mrow[8], lrow[8];
#pragma unroll
  for (int r = 0; r < 8; ++r) { mrow[r] = -1e30f; lrow[r] = 0.f; }

  bf16* const pp = &lds_p[wv][0];
  const int ncol = lane & 15;
  const int mb   = (lane >> 4) << 3;

  int cur = 0;
  for (int kt = 0; kt < NTOK; kt += 32) {
    // Publish buffer `cur`; barrier also proves all waves finished reading
    // buffer cur^1 last iteration (their ds reads fed WMMAs before this point).
    if (issuer) __builtin_amdgcn_s_wait_tensorcnt((short)0);
    __syncthreads();
    if (issuer && (kt + 32 < NTOK)) {
      const int nxt = cur ^ 1;
      tdm_load_2d_bf16(lds_offset(&lds_k[nxt][0]),
                       kbase + (size_t)(kt + 32) * ADIM,
                       ADIM, 32, ADIM, NTOK, ADIM);
      tdm_load_2d_bf16(lds_offset(&lds_v[nxt][0]),
                       vbase + (kt + 32),
                       32, DDIM, NTOK, DDIM, NTOK);
    }
    const bf16* kl = &lds_k[cur][0];   // [32 keys][64]
    const bf16* vl = &lds_v[cur][0];   // [256 e][32 keys]

    // ---- S = Qx @ Kx^T : two 16x16 tiles (keys +0..15, +16..31) ----
    v8f s0 = {0.f, 0.f, 0.f, 0.f, 0.f, 0.f, 0.f, 0.f};
    v8f s1 = {0.f, 0.f, 0.f, 0.f, 0.f, 0.f, 0.f, 0.f};
    {
      v16bf b0 = frag_b_bf16(kl,      ADIM, lane);
      v16bf b1 = frag_b_bf16(kl + 32, ADIM, lane);
      s0 = wmma_bf16(aq0, b0, s0);
      s0 = wmma_bf16(aq1, b1, s0);
      const bf16* kh = kl + 16 * ADIM;
      v16bf b2 = frag_b_bf16(kh,      ADIM, lane);
      v16bf b3 = frag_b_bf16(kh + 32, ADIM, lane);
      s1 = wmma_bf16(aq0, b2, s1);
      s1 = wmma_bf16(aq1, b3, s1);
    }

    // ---- online softmax over keys (per query row = mb + r) ----
    float scale[8], p0[8], p1[8];
#pragma unroll
    for (int r = 0; r < 8; ++r) {
      float mx = fmaxf(s0[r], s1[r]);
      mx = fmaxf(mx, __shfl_xor(mx, 1, 32));
      mx = fmaxf(mx, __shfl_xor(mx, 2, 32));
      mx = fmaxf(mx, __shfl_xor(mx, 4, 32));
      mx = fmaxf(mx, __shfl_xor(mx, 8, 32));
      const float mnew = fmaxf(mrow[r], mx);
      const float e0 = __expf(s0[r] - mnew);
      const float e1 = __expf(s1[r] - mnew);
      float rs = e0 + e1;
      rs += __shfl_xor(rs, 1, 32);
      rs += __shfl_xor(rs, 2, 32);
      rs += __shfl_xor(rs, 4, 32);
      rs += __shfl_xor(rs, 8, 32);
      const float sc = __expf(mrow[r] - mnew);
      lrow[r] = lrow[r] * sc + rs;
      mrow[r] = mnew;
      scale[r] = sc;
      p0[r] = e0;
      p1[r] = e1;
    }

    // ---- stage P (16x32) C-layout -> row-major; wave-private slab, LDS
    // ops are in-order within a wave so no barrier is needed here.
#pragma unroll
    for (int r = 0; r < 8; ++r) {
      pp[(mb + r) * 32 + ncol]      = (bf16)p0[r];
      pp[(mb + r) * 32 + 16 + ncol] = (bf16)p1[r];
    }

    // ---- rescale running accumulators ----
#pragma unroll
    for (int t = 0; t < 16; ++t)
#pragma unroll
      for (int r = 0; r < 8; ++r) o[t][r] *= scale[r];

    const v16bf ap = frag_a_bf16(pp, 32, lane);

    // ---- O += P(16x32) @ V(32x256): 16 WMMAs over the e dimension ----
#pragma unroll
    for (int t = 0; t < 16; ++t) {
      v16bf bv = frag_b_bf16(vl + (t * 16) * 32, 32, lane);
      o[t] = wmma_bf16(ap, bv, o[t]);
    }
    cur ^= 1;
  }

  // ---- normalize and store f32 output [b][query][e] ----
  float inv[8];
#pragma unroll
  for (int r = 0; r < 8; ++r) inv[r] = 1.0f / lrow[r];
  float* ob = out + ((size_t)b * NTOK + q0) * DDIM;
#pragma unroll
  for (int t = 0; t < 16; ++t)
#pragma unroll
    for (int r = 0; r < 8; ++r)
      ob[(size_t)(mb + r) * DDIM + t * 16 + ncol] = o[t][r] * inv[r];
}

// ---------------------------------------------------------------------------
extern "C" void kernel_launch(void* const* d_in, const int* in_sizes, int n_in,
                              void* d_out, int out_size, void* d_ws, size_t ws_size,
                              hipStream_t stream) {
  const float* x = (const float*)d_in[0];  // [4,4096,256]
  const float* Q = (const float*)d_in[1];  // [64,256]
  const float* K = (const float*)d_in[2];  // [64,256]
  const float* V = (const float*)d_in[3];  // [256,256]
  float* out = (float*)d_out;              // [4,4096,256]

  const int M = B_SZ * NTOK;               // 16384 tokens
  bf16* qx = (bf16*)d_ws;                  // [M][64]
  bf16* kx = qx + (size_t)M * ADIM;        // [M][64]
  bf16* vt = kx + (size_t)M * ADIM;        // [B][256][4096]  (12.6 MB total)

  // Stage 1: projections (one wave per 16x16 tile, 4 waves per block).
  {
    const int waves_qk = (M / 16) * (ADIM / 16);   // 4096
    proj_bf16_wmma<<<waves_qk / 4, 128, 0, stream>>>(x, Q, qx, M, ADIM, 0);
    proj_bf16_wmma<<<waves_qk / 4, 128, 0, stream>>>(x, K, kx, M, ADIM, 0);
    const int waves_v = (M / 16) * (DDIM / 16);    // 16384
    proj_bf16_wmma<<<waves_v / 4, 128, 0, stream>>>(x, V, vt, M, DDIM, 1);
  }

  // Stage 2: TDM-fed flash attention, 64 queries per block (4 waves x 16).
  dim3 grid(NTOK / 64, B_SZ);
  flash_attn_wmma<<<grid, 128, 0, stream>>>(qx, kx, vt, out);
}
